// Recurrent_43104291782798
// MI455X (gfx1250) — compile-verified, matrix operand order checked by reading, not run on hardware
//
#include <hip/hip_runtime.h>
#include <math.h>

#define BSZ  32
#define TLEN 4096
#define DK   128
#define HN   128
#define TH   (TLEN * HN)

typedef __attribute__((ext_vector_type(16))) _Float16 v16h;
typedef __attribute__((ext_vector_type(8)))  _Float16 v8h;
typedef __attribute__((ext_vector_type(8)))  float    v8f;

static __device__ __forceinline__ float fast_tanh(float x) {
#if __has_builtin(__builtin_amdgcn_tanhf)
  return __builtin_amdgcn_tanhf(x);
#elif __has_builtin(__builtin_amdgcn_tanh_f32)
  return __builtin_amdgcn_tanh_f32(x);
#else
  return tanhf(x);
#endif
}

// ---------------------------------------------------------------------------
// Phase 1: xproj[b,t,:] = x[b,t,:] @ Wx + bias  -> written into d_out.
// Rows R = B*T = 131072, K = 128, N = 128. One wave = one 16x16 tile,
// 8 m-tiles per wave task to amortize the Wx fragment loads.
// ---------------------------------------------------------------------------
__global__ __launch_bounds__(256) void rnn_xproj(const float* __restrict__ x,
                                                 const float* __restrict__ Wx,
                                                 const float* __restrict__ bias,
                                                 float* __restrict__ out) {
  const int lane  = threadIdx.x & 31;
  const int wave  = threadIdx.x >> 5;           // 0..7
  const int lidx  = lane & 15;
  const int lhalf = lane >> 4;                  // 0/1
  const int task  = blockIdx.x * 8 + wave;      // 8192 wave tasks
  const int n_tile  = task & 7;                 // 0..7
  const int m_block = task >> 3;                // 0..1023
  const int n = n_tile * 16 + lidx;             // output column

  // B operand (Wx) fragments: lane holds column n; frag f covers K=32f..32f+31.
  v16h bf[4];
#pragma unroll
  for (int f = 0; f < 4; ++f) {
    const int k0 = 32 * f + 8 * lhalf;
#pragma unroll
    for (int j = 0; j < 8; ++j) {
      bf[f][j]     = (_Float16)Wx[(k0 + j)      * HN + n];
      bf[f][j + 8] = (_Float16)Wx[(k0 + 16 + j) * HN + n];
    }
  }
  const float bv = bias[n];

#pragma unroll 1
  for (int mi = 0; mi < 8; ++mi) {
    const long m_tile = (long)m_block * 8 + mi;            // 0..8191
    const float* __restrict__ xrow = x + (m_tile * 16 + lidx) * (long)DK;

    v16h af[4];
#pragma unroll
    for (int f = 0; f < 4; ++f) {
      const int k0 = 32 * f + 8 * lhalf;
#pragma unroll
      for (int j = 0; j < 8; ++j) {
        af[f][j]     = (_Float16)xrow[k0 + j];
        af[f][j + 8] = (_Float16)xrow[k0 + 16 + j];
      }
    }

    v8f acc;
#pragma unroll
    for (int r = 0; r < 8; ++r) acc[r] = bv;               // fuse bias into C
#pragma unroll
    for (int f = 0; f < 4; ++f)
      acc = __builtin_amdgcn_wmma_f32_16x16x32_f16(
          false, af[f], false, bf[f], (short)0, acc, false, false);

    float* __restrict__ orow = out + (m_tile * 16 + 8 * lhalf) * (long)HN + n;
#pragma unroll
    for (int r = 0; r < 8; ++r) orow[(long)r * HN] = acc[r];
  }
}

// ---------------------------------------------------------------------------
// Phase 2: sequential scan h_t = tanh(xproj_t + h_{t-1} @ Wh), in-place on
// d_out. One workgroup, 16 waves: wave (mt, nt) owns 16x16 tile of the 32x128
// state. Wh fragments live in registers for the whole loop; h lives in a
// double-buffered LDS f16 array; xproj tile for t+1 prefetched during step t.
// ---------------------------------------------------------------------------
__global__ __launch_bounds__(512) void rnn_scan(const float* __restrict__ Wh,
                                                const float* __restrict__ h0,
                                                float* __restrict__ io) {
  __shared__ _Float16 hbuf[2][BSZ][HN];

  const int lane  = threadIdx.x & 31;
  const int wave  = threadIdx.x >> 5;   // 0..15
  const int lidx  = lane & 15;
  const int lhalf = lane >> 4;
  const int mt = wave >> 3;             // 0/1 : which 16 batch rows
  const int nt = wave & 7;              // 0..7: which 16 hidden cols
  const int n  = nt * 16 + lidx;

  // Loop-invariant B operand: Wh columns for this wave's N tile.
  v16h wf[4];
#pragma unroll
  for (int f = 0; f < 4; ++f) {
    const int k0 = 32 * f + 8 * lhalf;
#pragma unroll
    for (int j = 0; j < 8; ++j) {
      wf[f][j]     = (_Float16)Wh[(k0 + j)      * HN + n];
      wf[f][j + 8] = (_Float16)Wh[(k0 + 16 + j) * HN + n];
    }
  }

  // h_{-1} = h0 into LDS buffer 0.
  for (int i = threadIdx.x; i < BSZ * HN; i += 512)
    (&hbuf[0][0][0])[i] = (_Float16)h0[i];
  __syncthreads();

  // C/D layout: VGPR r -> batch row b0 + r, column n.
  const int b0 = mt * 16 + 8 * lhalf;
  float* __restrict__ p = io + (long)b0 * TH + n;   // + r*TH + t*HN

  // Prefetch xproj tile for t = 0.
  v8f cnext;
#pragma unroll
  for (int r = 0; r < 8; ++r) cnext[r] = p[(long)r * TH];

  int cur = 0;
#pragma unroll 1
  for (int t = 0; t < TLEN; ++t) {
    v8f acc = cnext;                                // C := xproj tile (fused add)

    // Prefetch next step's tile (independent of the recurrence).
    if (t + 1 < TLEN) {
      const long off = (long)(t + 1) * HN;
#pragma unroll
      for (int r = 0; r < 8; ++r) cnext[r] = p[(long)r * TH + off];
    }

    // A operand: rows M = mt*16 + lidx of h_{t-1} from LDS (16B-aligned b128s).
    const _Float16* hrow = &hbuf[cur][mt * 16 + lidx][0];
    v16h af[4];
#pragma unroll
    for (int f = 0; f < 4; ++f) {
      const int k0 = 32 * f + 8 * lhalf;
      v8h lo = *(const v8h*)(hrow + k0);
      v8h hi = *(const v8h*)(hrow + k0 + 16);
#pragma unroll
      for (int j = 0; j < 8; ++j) { af[f][j] = lo[j]; af[f][j + 8] = hi[j]; }
    }

#pragma unroll
    for (int f = 0; f < 4; ++f)
      acc = __builtin_amdgcn_wmma_f32_16x16x32_f16(
          false, af[f], false, wf[f], (short)0, acc, false, false);

#pragma unroll
    for (int r = 0; r < 8; ++r) acc[r] = fast_tanh(acc[r]);

    // h_t out to global (overwrites xproj slot) ...
    const long toff = (long)t * HN;
#pragma unroll
    for (int r = 0; r < 8; ++r) p[(long)r * TH + toff] = acc[r];

    // ... and to the other LDS buffer for step t+1.
    const int nxt = cur ^ 1;
#pragma unroll
    for (int r = 0; r < 8; ++r) hbuf[nxt][b0 + r][n] = (_Float16)acc[r];

    __syncthreads();
    cur = nxt;
  }
}

extern "C" void kernel_launch(void* const* d_in, const int* in_sizes, int n_in,
                              void* d_out, int out_size, void* d_ws, size_t ws_size,
                              hipStream_t stream) {
  (void)in_sizes; (void)n_in; (void)out_size; (void)d_ws; (void)ws_size;
  const float* x  = (const float*)d_in[0];   // [B,T,D]
  const float* Wx = (const float*)d_in[1];   // [D,H]
  const float* Wh = (const float*)d_in[2];   // [H,H]
  const float* b  = (const float*)d_in[3];   // [H]
  const float* h0 = (const float*)d_in[4];   // [B,H]
  float* out = (float*)d_out;                // [B,T,H]

  // Phase 1: xproj into d_out. 8192 wave tasks / 8 waves per 256-thread block.
  rnn_xproj<<<dim3(1024), dim3(256), 0, stream>>>(x, Wx, b, out);
  // Phase 2: sequential scan, single workgroup of 16 waves.
  rnn_scan<<<dim3(1), dim3(512), 0, stream>>>(Wh, h0, out);
}
